// MiniGRUConv2d8_3650722201911
// MI455X (gfx1250) — matrix operand order repaired and checked by reading, compiled, not run on hardware
//
#include <hip/hip_runtime.h>
#include <hip/hip_bf16.h>
#include <math.h>

typedef __attribute__((ext_vector_type(16))) _Float16 v16h;
typedef __attribute__((ext_vector_type(8)))  _Float16 v8h;
typedef __attribute__((ext_vector_type(8)))  float    v8f;

#define BB   4
#define CIN  64
#define HH   128
#define WW   128
#define CDIR 64
#define NCH  512      // per-conv output channels
#define NTOT 1536     // 3 convs concatenated
#define KTOT 576      // 9*64

// ---- gfx1250 async global->LDS path (guarded; falls back to plain copy) ----
#if defined(__gfx1250__) && __has_builtin(__builtin_amdgcn_global_load_async_to_lds_b128)
#define HAS_ASYNC_LDS 1
// b128 variant takes pointers to a 16-byte int vector (per compiler diagnostic):
typedef int v4i_vs __attribute__((__vector_size__(16)));
typedef __attribute__((address_space(1))) v4i_vs g_v4i;   // global (__device__)
typedef __attribute__((address_space(3))) v4i_vs l_v4i;   // LDS (__shared__)
#define TO_GLOBAL(p) ((g_v4i*)(uintptr_t)(p))
#define TO_LDS(p)    ((l_v4i*)(uint32_t)(uintptr_t)(p))
#if __has_builtin(__builtin_amdgcn_s_wait_asynccnt)
#define WAIT_ASYNC() __builtin_amdgcn_s_wait_asynccnt(0)
#else
#define WAIT_ASYNC() asm volatile("s_wait_asynccnt 0x0" ::: "memory")
#endif
#else
#define HAS_ASYNC_LDS 0
#define WAIT_ASYNC()
#endif

// ---------------------------------------------------------------------------
// f32 -> f16 input conversion
// ---------------------------------------------------------------------------
__global__ void cvt_x_kernel(const float* __restrict__ x, _Float16* __restrict__ xh, int n) {
  int i = blockIdx.x * blockDim.x + threadIdx.x;
  if (i < n) xh[i] = (_Float16)x[i];
}

// Repack (Cout,Cin,3,3) f32 weights of 3 convs into one f16 matrix [n][khw][c]
// so the GEMM K ordering is (kh,kw,c): K-chunks of 32 stay inside one (kh,kw).
__global__ void cvt_w_kernel(const float* __restrict__ zw, const float* __restrict__ hw,
                             const float* __restrict__ sw, _Float16* __restrict__ wh, int n) {
  int i = blockIdx.x * blockDim.x + threadIdx.x;
  if (i >= n) return;
  int c   = i & 63;
  int khw = (i >> 6) % 9;
  int nn  = i / KTOT;
  int t   = nn >> 9;       // which conv
  int nl  = nn & 511;      // channel within conv
  const float* src = (t == 0) ? zw : (t == 1) ? hw : sw;
  wh[i] = (_Float16)src[(size_t)(nl * CIN + c) * 9 + khw];  // (Cout,Cin,kh,kw)
}

// ---------------------------------------------------------------------------
// Fused implicit-GEMM conv for all three convolutions:
//   M = B*H*W (tile = one (b,h) row of 128 w), N = 1536, K = 576 (chunks of 32)
// Double-buffered LDS, async global->LDS weight staging, software pipeline.
// Epilogue: +bias, sigmoid for z (n<512) and s (n>=1024), f32 store.
// ---------------------------------------------------------------------------
__global__ __launch_bounds__(256) void conv_wmma_kernel(
    const _Float16* __restrict__ xh, const _Float16* __restrict__ wh,
    const float* __restrict__ zb, const float* __restrict__ hbias, const float* __restrict__ sbias,
    float* __restrict__ zs, float* __restrict__ hsb, float* __restrict__ ssb) {
  __shared__ _Float16 Alds[2][128][40];   // 128 m x 32 k, padded rows (16B-aligned frags)
  __shared__ _Float16 Blds[2][128][40];   // 128 n x 32 k

  const int mt   = blockIdx.x;         // 0..511 -> (b,h)
  const int nt   = blockIdx.y;         // 0..11  -> 128-wide n tile
  const int b    = mt >> 7;
  const int h    = mt & 127;
  const int tid  = threadIdx.x;
  const int lane = tid & 31;
  const int wave = tid >> 5;
  const int wm   = wave & 3;           // 4 waves along M (32 rows each)
  const int wn   = wave >> 2;          // 2 waves along N (64 cols each)
  const int lm   = lane & 15;
  const int kh2  = lane >> 4;          // lane half (K/M split per ISA layout)

  v8f acc[2][4] = {};

  const int cc  = tid & 31;            // A loader: channel within chunk
  const int grp = tid >> 5;            // A loader: 16-wide w group
  const int nl  = tid >> 1;            // B loader: n row
  const int seg = tid & 1;             // B loader: 16-half segment

  auto load_tiles = [&](int kc, int buf) {
    const int khw = kc >> 1;
    const int kh  = khw / 3, kw = khw % 3;
    const int c0  = (kc & 1) << 5;

    // ---- B tile: weights [128 n][32 c] (contiguous in wh) ----
    const _Float16* srcp =
        wh + ((size_t)(nt * 128 + nl) * 9 + khw) * 64 + c0 + seg * 16;
#if HAS_ASYNC_LDS
    __builtin_amdgcn_global_load_async_to_lds_b128(
        TO_GLOBAL(srcp),     TO_LDS(&Blds[buf][nl][seg * 16]),     0, 0);
    __builtin_amdgcn_global_load_async_to_lds_b128(
        TO_GLOBAL(srcp + 8), TO_LDS(&Blds[buf][nl][seg * 16 + 8]), 0, 0);
#else
    *(v8h*)&Blds[buf][nl][seg * 16]     = *(const v8h*)(srcp);
    *(v8h*)&Blds[buf][nl][seg * 16 + 8] = *(const v8h*)(srcp + 8);
#endif

    // ---- A tile: im2col, x[b][c0+cc][h+kh-1][w+kw-1] for w=0..127 ----
    {
      const int  hh  = h + kh - 1;
      const bool hok = (hh >= 0) && (hh < HH);
      const _Float16* xrow =
          xh + ((size_t)(b * CIN + c0 + cc) * HH + (hok ? hh : 0)) * WW;
      #pragma unroll
      for (int u = 0; u < 16; ++u) {
        const int wl   = grp * 16 + u;
        const int wsrc = wl + kw - 1;
        _Float16  v    = (_Float16)0.f;
        if (hok && wsrc >= 0 && wsrc < WW) v = xrow[wsrc];
        Alds[buf][wl][cc] = v;
      }
    }
    // prefetch the next chunk's x row (weights already stream via async DMA)
    if (kc + 1 < 18) {
      const int khw2 = (kc + 1) >> 1;
      const int kh_2 = khw2 / 3;
      const int c02  = ((kc + 1) & 1) << 5;
      const int hh2  = h + kh_2 - 1;
      if (hh2 >= 0 && hh2 < HH)
        __builtin_prefetch(xh + ((size_t)(b * CIN + c02 + cc) * HH + hh2) * WW, 0, 1);
    }
  };

  // -------- software pipeline over 18 K-chunks --------
  load_tiles(0, 0);
  WAIT_ASYNC();
  __syncthreads();

  for (int kc = 0; kc < 18; ++kc) {
    const int buf = kc & 1;
    if (kc + 1 < 18) load_tiles(kc + 1, buf ^ 1);   // overlaps with WMMAs below

    // ---- build fragments (ISA 16-bit A/B lane layouts) + WMMA ----
    v16h af[2];
    #pragma unroll
    for (int i = 0; i < 2; ++i) {
      const int m = wm * 32 + i * 16 + lm;
      v8h lo = *(const v8h*)&Alds[buf][m][8 * kh2];        // K 0..7   (or 8..15)
      v8h hi = *(const v8h*)&Alds[buf][m][16 + 8 * kh2];   // K 16..23 (or 24..31)
      af[i] = __builtin_shufflevector(lo, hi, 0,1,2,3,4,5,6,7,8,9,10,11,12,13,14,15);
    }
    v16h bf[4];
    #pragma unroll
    for (int j = 0; j < 4; ++j) {
      const int n = wn * 64 + j * 16 + lm;
      v8h lo = *(const v8h*)&Blds[buf][n][8 * kh2];        // K 0..7   (or 8..15)
      v8h hi = *(const v8h*)&Blds[buf][n][16 + 8 * kh2];   // K 16..23 (or 24..31)
      bf[j] = __builtin_shufflevector(lo, hi, 0,1,2,3,4,5,6,7,8,9,10,11,12,13,14,15);
    }
    #pragma unroll
    for (int i = 0; i < 2; ++i)
      #pragma unroll
      for (int j = 0; j < 4; ++j)
        acc[i][j] = __builtin_amdgcn_wmma_f32_16x16x32_f16(
            false, af[i], false, bf[j], (short)0, acc[i][j], false, false);

    WAIT_ASYNC();      // next chunk's async weight tiles landed
    __syncthreads();   // and everyone is done reading buf before it is reused
  }

  // ---- epilogue: bias + sigmoid(z,s), store f32 ----
  #pragma unroll
  for (int j = 0; j < 4; ++j) {
    const int n     = nt * 128 + wn * 64 + j * 16 + lm;
    const int which = n >> 9;                  // 0=z, 1=h, 2=s (uniform per 16-block)
    const int ch    = n - (which << 9);
    const float bias = (which == 0) ? zb[ch] : (which == 1) ? hbias[ch] : sbias[ch];
    float* base = (which == 0) ? zs : (which == 1) ? hsb : ssb;
    float* outp = base + ((size_t)(b * NCH + ch) * HH + h) * WW;
    #pragma unroll
    for (int i = 0; i < 2; ++i) {
      const int mb = wm * 32 + i * 16 + 8 * kh2;     // C layout: M = r + 8*lanehalf
      #pragma unroll
      for (int r = 0; r < 8; ++r) {
        float v = acc[i][j][r] + bias;
        if (which != 1) v = 1.0f / (1.0f + __expf(-v));
        outp[mb + r] = v;
      }
    }
  }
}

// ---------------------------------------------------------------------------
// 8-direction GRU scans; results overwrite h buffer in place.
// grid = (B*64, 8); block = 128
// ---------------------------------------------------------------------------
__global__ __launch_bounds__(128) void scan_kernel(
    const float* __restrict__ zs, float* __restrict__ hsb,
    const float* __restrict__ i0, const float* __restrict__ i1,
    const float* __restrict__ i2, const float* __restrict__ i3,
    const float* __restrict__ i4, const float* __restrict__ i5,
    const float* __restrict__ i6, const float* __restrict__ i7) {
  __shared__ float cbuf[130];
  const int dir = blockIdx.y;
  const int bc  = blockIdx.x;
  const int b   = bc >> 6, c = bc & 63;
  const int ch  = dir * 64 + c;
  const size_t off = (size_t)(b * NCH + ch) * HH * WW;
  const float* zp = zs + off;
  float*       hp = hsb + off;
  const float* iv;
  switch (dir) {
    case 0: iv = i0; break; case 1: iv = i1; break;
    case 2: iv = i2; break; case 3: iv = i3; break;
    case 4: iv = i4; break; case 5: iv = i5; break;
    case 6: iv = i6; break; default: iv = i7; break;
  }
  const float init = iv[c];
  const int t = threadIdx.x;
  float carry = init;

  if (dir == 0) {                                   // top -> down, t = w
    for (int hh = 0; hh < HH; ++hh) {
      const int idx = hh * WW + t;
      const float z = zp[idx], x = hp[idx];
      carry = z * x + (1.f - z) * carry;
      hp[idx] = carry;
    }
  } else if (dir == 1) {                            // down -> top
    for (int hh = HH - 1; hh >= 0; --hh) {
      const int idx = hh * WW + t;
      const float z = zp[idx], x = hp[idx];
      carry = z * x + (1.f - z) * carry;
      hp[idx] = carry;
    }
  } else if (dir == 2) {                            // left -> right, t = h
    const float* zr = zp + t * WW;
    float*       hr = hp + t * WW;
    for (int w = 0; w < WW; ++w) {
      const float z = zr[w], x = hr[w];
      carry = z * x + (1.f - z) * carry;
      hr[w] = carry;
    }
  } else if (dir == 3) {                            // right -> left
    const float* zr = zp + t * WW;
    float*       hr = hp + t * WW;
    for (int w = WW - 1; w >= 0; --w) {
      const float z = zr[w], x = hr[w];
      carry = z * x + (1.f - z) * carry;
      hr[w] = carry;
    }
  } else {                                          // diagonal scans, t = w
    const bool rev = (dir >= 6);
    const bool shp = (dir == 4) || (dir == 6);      // +1: use carry[j-1]
    for (int s = 0; s < HH; ++s) {
      const int hh = rev ? (HH - 1 - s) : s;
      cbuf[t + 1] = carry;
      if (t == 0)   cbuf[0]   = init;
      if (t == 127) cbuf[129] = init;
      __syncthreads();
      const float sh = shp ? cbuf[t] : cbuf[t + 2];
      const int idx = hh * WW + t;
      const float z = zp[idx], x = hp[idx];
      carry = z * x + (1.f - z) * sh;
      hp[idx] = carry;
      __syncthreads();
    }
  }
}

// ---------------------------------------------------------------------------
// out[b][c][h][w] = sum_k ss[b][k*64+c][h][w] * hscan[b][k*64+c][h][w]
// ---------------------------------------------------------------------------
__global__ void combine_kernel(const float* __restrict__ hsb,
                               const float* __restrict__ ssb,
                               float* __restrict__ out, int n) {
  const int i = blockIdx.x * blockDim.x + threadIdx.x;
  if (i >= n) return;
  const int hw = i & (HH * WW - 1);
  const int bc = i >> 14;
  const int b = bc >> 6, c = bc & 63;
  float s = 0.f;
  #pragma unroll
  for (int k = 0; k < 8; ++k) {
    const size_t idx = ((size_t)(b * NCH + k * 64 + c) << 14) + hw;
    s += ssb[idx] * hsb[idx];
  }
  out[i] = s;
}

// ---------------------------------------------------------------------------
extern "C" void kernel_launch(void* const* d_in, const int* in_sizes, int n_in,
                              void* d_out, int out_size, void* d_ws, size_t ws_size,
                              hipStream_t stream) {
  (void)in_sizes; (void)n_in; (void)out_size; (void)ws_size;
  const float* xs  = (const float*)d_in[0];
  const float* z_w = (const float*)d_in[1];
  const float* z_b = (const float*)d_in[2];
  const float* h_w = (const float*)d_in[3];
  const float* h_b = (const float*)d_in[4];
  const float* s_w = (const float*)d_in[5];
  const float* s_b = (const float*)d_in[6];

  char* ws = (char*)d_ws;
  const size_t ZS = (size_t)BB * NCH * HH * WW * sizeof(float);   // 128 MiB each
  float*    zs  = (float*)(ws);
  float*    hsb = (float*)(ws + ZS);
  float*    ssb = (float*)(ws + 2 * ZS);
  _Float16* xh  = (_Float16*)(ws + 3 * ZS);
  _Float16* wh  = (_Float16*)(ws + 3 * ZS + (size_t)BB * CIN * HH * WW * sizeof(_Float16));

  { int n = BB * CIN * HH * WW;
    cvt_x_kernel<<<(n + 255) / 256, 256, 0, stream>>>(xs, xh, n); }
  { int n = NTOT * KTOT;
    cvt_w_kernel<<<(n + 255) / 256, 256, 0, stream>>>(z_w, h_w, s_w, wh, n); }

  conv_wmma_kernel<<<dim3(BB * HH, NTOT / 128), 256, 0, stream>>>(
      xh, wh, z_b, h_b, s_b, zs, hsb, ssb);

  scan_kernel<<<dim3(BB * CDIR, 8), 128, 0, stream>>>(
      zs, hsb,
      (const float*)d_in[7],  (const float*)d_in[8],
      (const float*)d_in[9],  (const float*)d_in[10],
      (const float*)d_in[11], (const float*)d_in[12],
      (const float*)d_in[13], (const float*)d_in[14]);

  { int n = BB * CDIR * HH * WW;
    combine_kernel<<<(n + 255) / 256, 256, 0, stream>>>(hsb, ssb, (float*)d_out, n); }
}
